// graph_convolution_5471788335183
// MI455X (gfx1250) — matrix operand order repaired and checked by reading, not compiled, not run on hardware
//
#include <hip/hip_runtime.h>
#include <hip/hip_bf16.h>

typedef __attribute__((ext_vector_type(16))) __bf16 v16bf;
typedef __attribute__((ext_vector_type(8)))  float  v8f;

#define NNODES 50000
#define ROW_TILES (NNODES / 16)
static_assert(NNODES % 16 == 0, "row tiling requires N % 16 == 0");

// ---------------------------------------------------------------------------
// Zero-fill (float4 granularity, grid-stride)
// ---------------------------------------------------------------------------
__global__ void zero_f4(float* __restrict__ p, long n4) {
    long i = (long)blockIdx.x * blockDim.x + threadIdx.x;
    long stride = (long)gridDim.x * blockDim.x;
    float4 z = {0.f, 0.f, 0.f, 0.f};
    for (; i < n4; i += stride) ((float4*)p)[i] = z;
}

// ---------------------------------------------------------------------------
// Prepack fp32 weight [K, Ncols] into the CDNA5 WMMA bf16 B-operand layout:
//   dst[((colTile*(K/32) + chunk)*32 + lane)*16 + c]
//     = W[(chunk*32 + (lane<16 ? 0 : 8) + (c&7) + (c<8 ? 0 : 16)) * Ncols + colTile*16 + (lane&15)]
// so the GEMM inner loop fetches one contiguous v16bf (32B) per lane.
// ---------------------------------------------------------------------------
__global__ void prepack_w(const float* __restrict__ W, __bf16* __restrict__ dst,
                          int K, int Ncols) {
    int t = blockIdx.x * blockDim.x + threadIdx.x;
    int total = K * Ncols;
    if (t >= total) return;
    int c       = t & 15;
    int lane    = (t >> 4) & 31;
    int chunkId = t >> 9;
    int kchunks = K >> 5;
    int chunk   = chunkId % kchunks;
    int colTile = chunkId / kchunks;
    int n  = colTile * 16 + (lane & 15);
    int kk = chunk * 32 + ((lane < 16) ? 0 : 8) + (c & 7) + ((c < 8) ? 0 : 16);
    dst[t] = (__bf16)W[kk * Ncols + n];
}

// ---------------------------------------------------------------------------
// SpMM scatter: one wave per edge. Lane l handles feature cols [4l, 4l+3].
// Node matrices (<=51 MB) are L2-resident on the 192 MB L2, so both the
// random gather and the f32 atomic scatter run at L2 bandwidth.
// ---------------------------------------------------------------------------
__global__ void scatter_add(const int*   __restrict__ esrc,
                            const int*   __restrict__ edst,
                            const float* __restrict__ ew,
                            const float* __restrict__ x, long ldX,
                            float* __restrict__ acc, int E) {
    int gw   = (blockIdx.x * blockDim.x + threadIdx.x) >> 5;
    int lane = threadIdx.x & 31;
    if (gw >= E) return;
    int   s  = esrc[gw];
    int   d  = edst[gw];
    float wt = ew[gw];
    const float4 v = *(const float4*)(x + (long)s * ldX + lane * 4);
    float* o = acc + (long)d * 128 + lane * 4;
    atomicAdd(o + 0, wt * v.x);
    atomicAdd(o + 1, wt * v.y);
    atomicAdd(o + 2, wt * v.z);
    atomicAdd(o + 3, wt * v.w);
}

// ---------------------------------------------------------------------------
// Register-blocked GEMM: out[N, 16*NCT](f32, stride ldOut) =
//   [relu]( act[N, 32*KCH](f32, stride ldA) @ Wpacked(bf16) + bias )
// One wave owns a full 16-row x (16*NCT)-col output stripe: per K-chunk the
// A tile is loaded+converted ONCE and reused by NCT v_wmma_f32_16x16x32_bf16
// ops against NCT prepacked B tiles (NCT independent v8f accumulators).
// KCH/NCT are compile-time so B offsets become immediates.
// ---------------------------------------------------------------------------
template <int KCH, int NCT>
__global__ void gemm_bf16_wmma(const float* __restrict__ act, int ldA,
                               const __bf16* __restrict__ Wp,
                               const float* __restrict__ bias,
                               float* __restrict__ out, int ldOut,
                               int doRelu) {
    int gw   = (blockIdx.x * blockDim.x + threadIdx.x) >> 5;  // uniform per wave
    int lane = threadIdx.x & 31;
    if (gw >= ROW_TILES) return;  // wave-uniform: EXEC all-ones for WMMA

    int m  = lane & 15;           // A row / C column index within tile
    int hi = lane >> 4;           // half-wave select

    const float* arow = act + (long)(gw * 16 + m) * ldA;
    const v16bf* bp   = (const v16bf*)Wp + lane;

    v8f acc[NCT];
#pragma unroll
    for (int ct = 0; ct < NCT; ++ct) acc[ct] = (v8f){};

#pragma unroll
    for (int ch = 0; ch < KCH; ++ch) {
        int s0 = ch * 32 + hi * 8;
        // A lane layout: components 0..7  <- K = s0..s0+7
        //                components 8..15 <- K = s0+16..s0+23
        float4 f0 = *(const float4*)(arow + s0);
        float4 f1 = *(const float4*)(arow + s0 + 4);
        float4 f2 = *(const float4*)(arow + s0 + 16);
        float4 f3 = *(const float4*)(arow + s0 + 20);
        v16bf a;
        a[0]  = (__bf16)f0.x; a[1]  = (__bf16)f0.y; a[2]  = (__bf16)f0.z; a[3]  = (__bf16)f0.w;
        a[4]  = (__bf16)f1.x; a[5]  = (__bf16)f1.y; a[6]  = (__bf16)f1.z; a[7]  = (__bf16)f1.w;
        a[8]  = (__bf16)f2.x; a[9]  = (__bf16)f2.y; a[10] = (__bf16)f2.z; a[11] = (__bf16)f2.w;
        a[12] = (__bf16)f3.x; a[13] = (__bf16)f3.y; a[14] = (__bf16)f3.z; a[15] = (__bf16)f3.w;
#pragma unroll
        for (int ct = 0; ct < NCT; ++ct) {
            v16bf b = bp[(ct * KCH + ch) * 32];
            acc[ct] = __builtin_amdgcn_wmma_f32_16x16x32_bf16(
                /*neg_a=*/false, a, /*neg_b=*/false, b,
                /*c_mod=*/(short)0, acc[ct], /*reuse_a=*/false, /*reuse_b=*/false);
        }
    }

    // D layout: VGPR v, lanes 0-15 -> row v, col lane; lanes 16-31 -> row v+8
    int rbase = gw * 16 + hi * 8;
#pragma unroll
    for (int ct = 0; ct < NCT; ++ct) {
        int   n  = ct * 16 + m;
        float bv = bias ? bias[n] : 0.0f;
#pragma unroll
        for (int v = 0; v < 8; ++v) {
            float r = acc[ct][v] + bv;
            if (doRelu) r = fmaxf(r, 0.0f);
            out[(long)(rbase + v) * ldOut + n] = r;
        }
    }
}

// ---------------------------------------------------------------------------
// Launch pipeline (all on `stream`; scratch laid out in d_ws; buffers reused):
//   aggA:  N*128 f32   (spmm1 accum -> consumed -> t1 -> consumed -> hh)
//   aggB:  N*128 f32   (spmm2 accum)
//   h:     N*256 f32   (concat buffer: [selfc | conv1 | conv2])
//   bf16 prepacked weights after that.
// ---------------------------------------------------------------------------
extern "C" void kernel_launch(void* const* d_in, const int* in_sizes, int n_in,
                              void* d_out, int out_size, void* d_ws, size_t ws_size,
                              hipStream_t stream) {
    const float* features = (const float*)d_in[0];
    const int*   esrc     = (const int*)  d_in[1];
    const int*   edst     = (const int*)  d_in[2];
    const float* ew       = (const float*)d_in[3];
    const float* W1  = (const float*)d_in[4];
    const float* W2  = (const float*)d_in[5];
    const float* Wm1 = (const float*)d_in[6];
    const float* bm1 = (const float*)d_in[7];
    const float* Wm2 = (const float*)d_in[8];
    const float* bm2 = (const float*)d_in[9];
    const float* Wh1 = (const float*)d_in[10];
    const float* bh1 = (const float*)d_in[11];
    const float* Wh2 = (const float*)d_in[12];
    const float* bh2 = (const float*)d_in[13];
    const int E = in_sizes[1];

    float* wsf  = (float*)d_ws;
    float* aggA = wsf;                           // N*128
    float* aggB = wsf + (long)NNODES * 128;      // N*128
    float* h    = wsf + (long)NNODES * 256;      // N*256
    __bf16* bfw = (__bf16*)(wsf + (long)NNODES * 512);
    __bf16* W1p  = bfw;
    __bf16* W2p  = W1p  + 128 * 128;
    __bf16* Wm1p = W2p  + 128 * 64;
    __bf16* Wm2p = Wm1p + 128 * 128;
    __bf16* Wh1p = Wm2p + 128 * 64;
    __bf16* Wh2p = Wh1p + 256 * 128;

    // 1. zero both spmm accumulators (contiguous N*256 floats)
    zero_f4<<<4096, 256, 0, stream>>>(aggA, (long)NNODES * 256 / 4);

    // 2. prepack all six weight matrices to bf16 WMMA B layout
    auto pp = [&](const float* W, __bf16* dst, int K, int Ncols) {
        int tot = K * Ncols;
        prepack_w<<<(tot + 255) / 256, 256, 0, stream>>>(W, dst, K, Ncols);
    };
    pp(W1,  W1p,  128, 128);
    pp(W2,  W2p,  128, 64);
    pp(Wm1, Wm1p, 128, 128);
    pp(Wm2, Wm2p, 128, 64);
    pp(Wh1, Wh1p, 256, 128);
    pp(Wh2, Wh2p, 128, 64);

    const int gblocks = (ROW_TILES + 7) / 8;     // 8 waves / 256-thread block
    const int sblocks = (E + 7) / 8;             // one wave per edge

    // 3. agg1 = spmm(features)
    scatter_add<<<sblocks, 256, 0, stream>>>(esrc, edst, ew, features, 128, aggA, E);
    // 4. conv1 = relu(agg1 @ W1)           -> h cols [64,192)
    gemm_bf16_wmma<4, 8><<<gblocks, 256, 0, stream>>>(aggA, 128, W1p, nullptr, h + 64, 256, 1);
    // 5. t1 = relu(features @ Wm1 + bm1)   -> aggA (free after step 4)
    gemm_bf16_wmma<4, 8><<<gblocks, 256, 0, stream>>>(features, 128, Wm1p, bm1, aggA, 128, 1);
    // 6. selfc = t1 @ Wm2 + bm2            -> h cols [0,64)
    gemm_bf16_wmma<4, 4><<<gblocks, 256, 0, stream>>>(aggA, 128, Wm2p, bm2, h + 0, 256, 0);
    // 7. agg2 = spmm(conv1)  (gather conv1 straight out of h, stride 256)
    scatter_add<<<sblocks, 256, 0, stream>>>(esrc, edst, ew, h + 64, 256, aggB, E);
    // 8. conv2 = agg2 @ W2                 -> h cols [192,256)
    gemm_bf16_wmma<4, 4><<<gblocks, 256, 0, stream>>>(aggB, 128, W2p, nullptr, h + 192, 256, 0);
    // 9. hh = relu(h @ Wh1 + bh1)          -> aggA (reuse, t1 dead)
    gemm_bf16_wmma<8, 8><<<gblocks, 256, 0, stream>>>(h, 256, Wh1p, bh1, aggA, 128, 1);
    // 10. out = hh @ Wh2 + bh2             -> d_out [N,64]
    gemm_bf16_wmma<4, 4><<<gblocks, 256, 0, stream>>>(aggA, 128, Wh2p, bh2, (float*)d_out, 64, 0);
}
